// TSTransformerEncoder_19404662243647
// MI455X (gfx1250) — matrix operand order repaired and checked by reading, not compiled, not run on hardware
//
#include <hip/hip_runtime.h>
#include <hip/hip_bf16.h>
#include <math.h>

#define B_   32
#define S_   512
#define F_   64
#define D_   512
#define H_   8
#define L_   4
#define DFF_ 2048
#define HD_  64
#define NR_  (S_*B_)   // 16384 rows in (s,b) order

typedef _Float16 h16;
typedef __attribute__((ext_vector_type(16))) _Float16 v16h;
typedef __attribute__((ext_vector_type(8)))  float    v8f;

// ---------------------------------------------------------------------------
// WMMA fragment loaders (wave32, v_wmma_f32_16x16x32_f16 layouts per CDNA5 ISA)
// A: 16x32 f16, lane m = lane&15, K-half = (lane>>4)*8,
//    VGPR j holds K pair (j<4 ? 2j : 16+2(j-4)).
// B: 32x16 f16 from transposed weights Wt[N][K]; lane n = lane&15,
//    K-half = (lane>>4)*8, VGPR j holds K pair (j<4 ? 2j : 16+2(j-4)).
// C/D: 16x16 f32, lane n = lane&15, rows m = (lane>>4)*8 + j.
// ---------------------------------------------------------------------------
__device__ __forceinline__ v16h load_a16(const h16* A, int ld, int row0, int k0, int lane) {
  v16h a;
  int m  = row0 + (lane & 15);
  int kh = (lane >> 4) << 3;
  const h16* p = A + (size_t)m * ld + k0 + kh;
#pragma unroll
  for (int j = 0; j < 8; ++j) {
    int k = (j < 4) ? (2*j) : (16 + 2*(j-4));
    a[2*j]   = p[k];
    a[2*j+1] = p[k+1];
  }
  return a;
}

__device__ __forceinline__ v16h load_a32lds(const float* A, int ld, int row0, int k0, int lane) {
  v16h a;
  int m  = row0 + (lane & 15);
  int kh = (lane >> 4) << 3;
  const float* p = A + (size_t)m * ld + k0 + kh;
#pragma unroll
  for (int j = 0; j < 8; ++j) {
    int k = (j < 4) ? (2*j) : (16 + 2*(j-4));
    a[2*j]   = (h16)p[k];
    a[2*j+1] = (h16)p[k+1];
  }
  return a;
}

__device__ __forceinline__ v16h load_b16(const h16* Bt, int ld, int col0, int k0, int lane) {
  v16h b;
  int n  = col0 + (lane & 15);
  int kh = (lane >> 4) << 3;
  const h16* p = Bt + (size_t)n * ld + k0 + kh;
#pragma unroll
  for (int j = 0; j < 8; ++j) {
    int k = (j < 4) ? (2*j) : (16 + 2*(j-4));
    b[2*j]   = p[k];
    b[2*j+1] = p[k+1];
  }
  return b;
}

__device__ __forceinline__ float gelu_f(float x) {
  return 0.5f * x * (1.0f + erff(x * 0.70710678118654752f));
}

#define WMMA_F16(A_, B_x, C_) \
  __builtin_amdgcn_wmma_f32_16x16x32_f16(false, (A_), false, (B_x), (short)0, (C_), false, false)

__device__ __forceinline__ void store_tile(v8f acc, int r0, int c0,
                                           const float* bias, float* C32, h16* C16,
                                           int N, int act, float osc, int lane) {
  int n  = c0 + (lane & 15);
  int m0 = r0 + ((lane >> 4) << 3);
  float bv = bias ? bias[n] : 0.0f;
#pragma unroll
  for (int j = 0; j < 8; ++j) {
    float v = acc[j] + bv;
    if (act == 1)      v = fmaxf(v, 0.0f);
    else if (act == 2) v = gelu_f(v);
    v *= osc;
    size_t idx = (size_t)(m0 + j) * N + n;
    if (C32) C32[idx] = v;
    if (C16) C16[idx] = (h16)v;
  }
}

// ---------------------------------------------------------------------------
// Generic WMMA GEMM with 2x2 register blocking:
// block = 8 waves (2x4), block tile 64(M) x 128(N); each wave owns a 32x32
// output computed from 2 A-frags x 2 B-frags per K step -> 4 WMMA / 8 b128
// loads (2 loads per WMMA instead of 4).
// All launched N are multiples of 32, so col0 < N implies col0+16 < N: the
// inner loop is branch-free. The tile-bounds early-exit is scalarized via
// readfirstlane so it lowers to s_cbranch (EXEC stays all-ones for WMMA).
// C = act(A[M,K](f16) x Wt[N,K]^T(f16) + bias) * oscale
// act: 0=none 1=relu 2=gelu. C32/C16 optional outputs.
// ---------------------------------------------------------------------------
__global__ __launch_bounds__(256)
void gemm_ws(const h16* __restrict__ A, const h16* __restrict__ Bt,
             const float* __restrict__ bias, float* __restrict__ C32,
             h16* __restrict__ C16, int M, int N, int K, int act, float oscale) {
  int wave = threadIdx.x >> 5;
  int lane = threadIdx.x & 31;
  int row0 = blockIdx.y * 64  + (wave >> 2) * 32;
  int col0 = blockIdx.x * 128 + (wave & 3) * 32;
  // wave-uniform early exit, forced onto the scalar path
  if (__builtin_amdgcn_readfirstlane((row0 >= M || col0 >= N) ? 1 : 0)) return;

  v8f acc00 = {}, acc01 = {}, acc10 = {}, acc11 = {};
  for (int k = 0; k < K; k += 32) {
    v16h a0 = load_a16(A, K, row0,      k, lane);
    v16h a1 = load_a16(A, K, row0 + 16, k, lane);
    v16h b0 = load_b16(Bt, K, col0,      k, lane);
    v16h b1 = load_b16(Bt, K, col0 + 16, k, lane);
    acc00 = WMMA_F16(a0, b0, acc00);
    acc10 = WMMA_F16(a1, b0, acc10);
    acc01 = WMMA_F16(a0, b1, acc01);
    acc11 = WMMA_F16(a1, b1, acc11);
  }
  store_tile(acc00, row0,      col0,      bias, C32, C16, N, act, oscale, lane);
  store_tile(acc10, row0 + 16, col0,      bias, C32, C16, N, act, oscale, lane);
  store_tile(acc01, row0,      col0 + 16, bias, C32, C16, N, act, oscale, lane);
  store_tile(acc11, row0 + 16, col0 + 16, bias, C32, C16, N, act, oscale, lane);
}

// ---------------------------------------------------------------------------
// Fused attention for one (b,h) and a 32-row q chunk.
// Phase 1: scores = Q Kt^T * 1/sqrt(HD) into LDS (WMMA, Q-frags hoisted)
// Phase 2: masked softmax in LDS
// Phase 3: Y = P x V (WMMA), scattered back to y16[(s*B+b)*D + h*HD + d]
// Qh/Kh: [B*H, S, HD] f16 (rope applied). Vt: [B*H, HD, S] f16.
// ---------------------------------------------------------------------------
__global__ __launch_bounds__(256)
void attn_fused(const h16* __restrict__ Qh, const h16* __restrict__ Kh,
                const h16* __restrict__ Vt, const int* __restrict__ pm,
                h16* __restrict__ y16) {
  __shared__ float sc[32 * S_];   // 64 KB score/prob tile
  int bh = blockIdx.y;
  int b  = bh >> 3;               // bh / H
  int h  = bh & 7;
  int q0 = blockIdx.x * 32;
  const h16* Q  = Qh + (size_t)bh * S_ * HD_;
  const h16* Kt = Kh + (size_t)bh * S_ * HD_;   // [S, HD]: row = k-pos (== B^T layout)
  const h16* V  = Vt + (size_t)bh * HD_ * S_;   // [HD, S]

  int wave = threadIdx.x >> 5;
  int lane = threadIdx.x & 31;
  int wm   = wave >> 2;           // q sub-tile (0..1)

  // ---- phase 1: score tile 32 x 512 ; Q fragments loaded once per wave ----
  {
    int row0 = q0 + wm * 16;
    v16h aq0 = load_a16(Q, HD_, row0, 0,  lane);
    v16h aq1 = load_a16(Q, HD_, row0, 32, lane);
    for (int wn = (wave & 3); wn < S_ / 16; wn += 4) {
      int col0 = wn * 16;
      v16h b0 = load_b16(Kt, HD_, col0, 0,  lane);
      v16h b1 = load_b16(Kt, HD_, col0, 32, lane);
      v8f acc = {};
      acc = WMMA_F16(aq0, b0, acc);
      acc = WMMA_F16(aq1, b1, acc);
      int n  = col0 + (lane & 15);
      int m0 = wm * 16 + ((lane >> 4) << 3);
#pragma unroll
      for (int j = 0; j < 8; ++j)
        sc[(m0 + j) * S_ + n] = acc[j] * 0.125f;   // 1/sqrt(64)
    }
  }
  __syncthreads();

  // ---- phase 2: masked softmax, one thread per q row ----
  if (threadIdx.x < 32) {
    float* rowp = &sc[threadIdx.x * S_];
    const int* pmr = pm + b * S_;
    float mx = -1e30f;
    for (int k = 0; k < S_; ++k) {
      float v = pmr[k] ? -1e9f : rowp[k];   // keep where !padding_mask
      rowp[k] = v;
      mx = fmaxf(mx, v);
    }
    float sum = 0.0f;
    for (int k = 0; k < S_; ++k) {
      float e = expf(rowp[k] - mx);
      rowp[k] = e;
      sum += e;
    }
    float inv = 1.0f / sum;
    for (int k = 0; k < S_; ++k) rowp[k] *= inv;
  }
  __syncthreads();

  // ---- phase 3: Y = P x V ; tiles 2x4 = 8 waves ----
  {
    int row0l = wm * 16;
    int col0  = (wave & 3) * 16;
    v8f acc = {};
    for (int k = 0; k < S_; k += 32) {
      v16h a  = load_a32lds(sc, S_, row0l, k, lane);
      v16h bb = load_b16(V, S_, col0, k, lane);
      acc = WMMA_F16(a, bb, acc);
    }
    int n  = col0 + (lane & 15);
    int m0 = row0l + ((lane >> 4) << 3);
#pragma unroll
    for (int j = 0; j < 8; ++j) {
      int srow = q0 + m0 + j;   // q position s
      y16[((size_t)srow * B_ + b) * D_ + h * HD_ + n] = (h16)acc[j];
    }
  }
}

// ---------------------------------------------------------------------------
// Helper / elementwise kernels
// ---------------------------------------------------------------------------
__global__ void tconv(const float* __restrict__ W, h16* __restrict__ T, int K, int N) {
  int idx = blockIdx.x * blockDim.x + threadIdx.x;
  if (idx >= K * N) return;
  int k = idx / N, n = idx - k * N;
  T[(size_t)n * K + k] = (h16)W[idx];
}

__global__ void prep_x(const float* __restrict__ X, h16* __restrict__ x16) {
  int idx = blockIdx.x * blockDim.x + threadIdx.x;
  if (idx >= B_ * S_ * F_) return;
  int f = idx & (F_ - 1);
  int r = idx >> 6;
  int s = r & (S_ - 1);
  int b = r >> 9;
  x16[((size_t)s * B_ + b) * F_ + f] = (h16)X[idx];
}

__global__ void rope_heads(const h16* __restrict__ x16, h16* __restrict__ Xh) {
  int idx = blockIdx.x * blockDim.x + threadIdx.x;
  if (idx >= B_ * H_ * S_ * (HD_ / 2)) return;
  int i = idx & (HD_ / 2 - 1);
  int t = idx >> 5;
  int s = t & (S_ - 1); t >>= 9;
  int h = t & (H_ - 1);
  int b = t >> 3;
  size_t src = ((size_t)s * B_ + b) * D_ + h * HD_ + 2 * i;
  float xre = (float)x16[src], xim = (float)x16[src + 1];
  float theta = powf(10000.0f, -2.0f * (float)i / (float)HD_);
  float ang = (float)s * theta;
  float c = cosf(ang), sn = sinf(ang);
  float ev = xre * c - xim * sn;
  float od = xre * sn + xim * c;
  size_t dst = (((size_t)(b * H_ + h)) * S_ + s) * HD_ + 2 * i;
  Xh[dst]     = (h16)ev;
  Xh[dst + 1] = (h16)od;
}

__global__ void v_split_t(const h16* __restrict__ v16v, h16* __restrict__ Vt) {
  int idx = blockIdx.x * blockDim.x + threadIdx.x;
  if (idx >= B_ * H_ * S_ * HD_) return;
  int d = idx & (HD_ - 1);
  int t = idx >> 6;
  int s = t & (S_ - 1); t >>= 9;
  int h = t & (H_ - 1);
  int b = t >> 3;
  Vt[(((size_t)(b * H_ + h)) * HD_ + d) * S_ + s] =
      v16v[((size_t)s * B_ + b) * D_ + h * HD_ + d];
}

__global__ __launch_bounds__(256)
void add_ln(float* __restrict__ src, const float* __restrict__ y,
            const float* __restrict__ g, const float* __restrict__ be,
            h16* __restrict__ src16) {
  __shared__ float red[256];
  int row = blockIdx.x;
  int t = threadIdx.x;
  size_t base = (size_t)row * D_;
  float v0 = src[base + t]       + y[base + t];
  float v1 = src[base + t + 256] + y[base + t + 256];
  red[t] = v0 + v1;
  __syncthreads();
  for (int o = 128; o > 0; o >>= 1) { if (t < o) red[t] += red[t + o]; __syncthreads(); }
  float mean = red[0] * (1.0f / D_);
  __syncthreads();
  float d0 = v0 - mean, d1 = v1 - mean;
  red[t] = d0 * d0 + d1 * d1;
  __syncthreads();
  for (int o = 128; o > 0; o >>= 1) { if (t < o) red[t] += red[t + o]; __syncthreads(); }
  float inv = rsqrtf(red[0] * (1.0f / D_) + 1e-5f);
  float o0 = d0 * inv * g[t]       + be[t];
  float o1 = d1 * inv * g[t + 256] + be[t + 256];
  src[base + t]         = o0;  src16[base + t]       = (h16)o0;
  src[base + t + 256]   = o1;  src16[base + t + 256] = (h16)o1;
}

__global__ void gelu_permute(const float* __restrict__ src,
                             float* __restrict__ emb32, h16* __restrict__ emb16) {
  int idx = blockIdx.x * blockDim.x + threadIdx.x;
  if (idx >= NR_ * D_) return;
  int d = idx & (D_ - 1);
  int r = idx >> 9;
  int b = r & (B_ - 1);
  int s = r >> 5;
  float v = gelu_f(src[idx]);
  size_t drow = (size_t)b * S_ + s;
  emb32[drow * D_ + d] = v;
  emb16[drow * D_ + d] = (h16)v;
}

__global__ void pool_score(const float* __restrict__ emb32,
                           const float* __restrict__ Wpool,
                           const float* __restrict__ bpool,
                           float* __restrict__ ps) {
  int r = blockIdx.x * blockDim.x + threadIdx.x;
  if (r >= NR_) return;
  const float* e = emb32 + (size_t)r * D_;
  float acc = bpool[0];
  for (int d = 0; d < D_; ++d) acc += e[d] * Wpool[d];
  ps[r] = acc;
}

__global__ __launch_bounds__(256)
void pool_combine(const float* __restrict__ ps, const float* __restrict__ emb32,
                  float* __restrict__ pooled) {
  __shared__ float w[S_];
  __shared__ float red[256];
  int b = blockIdx.x;
  int t = threadIdx.x;
  float a0 = ps[b * S_ + t], a1 = ps[b * S_ + t + 256];
  red[t] = fmaxf(a0, a1);
  __syncthreads();
  for (int o = 128; o > 0; o >>= 1) { if (t < o) red[t] = fmaxf(red[t], red[t + o]); __syncthreads(); }
  float mx = red[0];
  __syncthreads();
  float e0 = expf(a0 - mx), e1 = expf(a1 - mx);
  w[t] = e0; w[t + 256] = e1;
  red[t] = e0 + e1;
  __syncthreads();
  for (int o = 128; o > 0; o >>= 1) { if (t < o) red[t] += red[t + o]; __syncthreads(); }
  float inv = 1.0f / red[0];
  float acc0 = 0.0f, acc1 = 0.0f;
  for (int s = 0; s < S_; ++s) {
    float ww = w[s] * inv;
    const float* e = emb32 + ((size_t)b * S_ + s) * D_;
    acc0 += ww * e[t];
    acc1 += ww * e[t + 256];
  }
  pooled[b * D_ + t]       = acc0;
  pooled[b * D_ + t + 256] = acc1;
}

__global__ void proj_k(const float* __restrict__ pooled, const float* __restrict__ Wproj,
                       const float* __restrict__ bproj, float* __restrict__ proj) {
  int idx = blockIdx.x * blockDim.x + threadIdx.x;
  if (idx >= B_ * (D_ / 2)) return;
  int j = idx & (D_ / 2 - 1);
  int b = idx >> 8;
  float acc = bproj[j];
  const float* p = pooled + (size_t)b * D_;
  for (int k = 0; k < D_; ++k) acc += p[k] * Wproj[(size_t)k * (D_ / 2) + j];
  proj[idx] = acc;
}

__global__ void bn_relu(const float* __restrict__ proj, const float* __restrict__ g,
                        const float* __restrict__ be, float* __restrict__ out) {
  int j = threadIdx.x;   // D/2 = 256
  float mean = 0.0f;
  for (int b = 0; b < B_; ++b) mean += proj[b * (D_ / 2) + j];
  mean *= (1.0f / B_);
  float var = 0.0f;
  for (int b = 0; b < B_; ++b) {
    float d = proj[b * (D_ / 2) + j] - mean;
    var += d * d;
  }
  var *= (1.0f / B_);
  float inv = rsqrtf(var + 1e-5f);
  for (int b = 0; b < B_; ++b) {
    float v = (proj[b * (D_ / 2) + j] - mean) * inv * g[j] + be[j];
    out[b * (D_ / 2) + j] = fmaxf(v, 0.0f);
  }
}

__global__ void fin_k(const float* __restrict__ pbn, const float* __restrict__ Wfin,
                      const float* __restrict__ bfin, float* __restrict__ out) {
  int idx = blockIdx.x * blockDim.x + threadIdx.x;
  if (idx >= B_ * (D_ / 4)) return;
  int j = idx & (D_ / 4 - 1);
  int b = idx >> 7;
  float acc = bfin[j];
  const float* p = pbn + (size_t)b * (D_ / 2);
  for (int k = 0; k < D_ / 2; ++k) acc += p[k] * Wfin[(size_t)k * (D_ / 4) + j];
  out[idx] = acc;
}

// ---------------------------------------------------------------------------
// Host launcher
// ---------------------------------------------------------------------------
static inline void launch_gemm(const h16* A, const h16* Bt, const float* bias,
                               float* C32, h16* C16, int M, int N, int K,
                               int act, float osc, hipStream_t s) {
  dim3 g((N + 127) / 128, (M + 63) / 64);
  gemm_ws<<<g, 256, 0, s>>>(A, Bt, bias, C32, C16, M, N, K, act, osc);
}

extern "C" void kernel_launch(void* const* d_in, const int* in_sizes, int n_in,
                              void* d_out, int out_size, void* d_ws, size_t ws_size,
                              hipStream_t stream) {
  const float* X     = (const float*)d_in[0];
  const int*   pm    = (const int*)  d_in[1];
  const float* Wp    = (const float*)d_in[2];
  const float* bp    = (const float*)d_in[3];
  const float* Wq    = (const float*)d_in[4];
  const float* bq    = (const float*)d_in[5];
  const float* Wk    = (const float*)d_in[6];
  const float* bk    = (const float*)d_in[7];
  const float* Wv    = (const float*)d_in[8];
  const float* bv    = (const float*)d_in[9];
  const float* Wo    = (const float*)d_in[10];
  const float* bo    = (const float*)d_in[11];
  const float* W1    = (const float*)d_in[12];
  const float* b1    = (const float*)d_in[13];
  const float* W2    = (const float*)d_in[14];
  const float* b2    = (const float*)d_in[15];
  const float* ln1g  = (const float*)d_in[16];
  const float* ln1b  = (const float*)d_in[17];
  const float* ln2g  = (const float*)d_in[18];
  const float* ln2b  = (const float*)d_in[19];
  const float* Wout  = (const float*)d_in[20];
  const float* bout  = (const float*)d_in[21];
  const float* Wpool = (const float*)d_in[22];
  const float* bpool = (const float*)d_in[23];
  const float* Wproj = (const float*)d_in[24];
  const float* bproj = (const float*)d_in[25];
  const float* bng   = (const float*)d_in[26];
  const float* bnb   = (const float*)d_in[27];
  const float* Wfin  = (const float*)d_in[28];
  const float* bfin  = (const float*)d_in[29];

  // ---- workspace bump allocator ----
  char* wsb = (char*)d_ws;
  size_t off = 0;
  auto alloc = [&](size_t bytes) -> void* {
    off = (off + 255) & ~(size_t)255;
    void* p = wsb + off;
    off += bytes;
    return p;
  };
  h16*   x16    = (h16*)  alloc((size_t)NR_ * F_  * 2);
  float* src32  = (float*)alloc((size_t)NR_ * D_  * 4);
  h16*   src16  = (h16*)  alloc((size_t)NR_ * D_  * 2);
  h16*   q16    = (h16*)  alloc((size_t)NR_ * D_  * 2);
  h16*   k16    = (h16*)  alloc((size_t)NR_ * D_  * 2);
  h16*   v16v   = (h16*)  alloc((size_t)NR_ * D_  * 2);
  h16*   Qh     = (h16*)  alloc((size_t)NR_ * D_  * 2);
  h16*   Kh     = (h16*)  alloc((size_t)NR_ * D_  * 2);
  h16*   Vt     = (h16*)  alloc((size_t)NR_ * D_  * 2);
  h16*   y16    = (h16*)  alloc((size_t)NR_ * D_  * 2);
  float* t32    = (float*)alloc((size_t)NR_ * D_  * 4);
  h16*   ff16   = (h16*)  alloc((size_t)NR_ * DFF_ * 2);
  float* emb32  = (float*)alloc((size_t)NR_ * D_  * 4);
  h16*   emb16  = (h16*)  alloc((size_t)NR_ * D_  * 2);
  float* pscore = (float*)alloc((size_t)NR_ * 4);
  float* pooled = (float*)alloc((size_t)B_ * D_ * 4);
  float* projb  = (float*)alloc((size_t)B_ * (D_/2) * 4);
  float* projbn = (float*)alloc((size_t)B_ * (D_/2) * 4);
  h16*   WpT    = (h16*)  alloc((size_t)D_ * F_ * 2);
  h16*   WqT    = (h16*)  alloc((size_t)L_ * D_ * D_ * 2);
  h16*   WkT    = (h16*)  alloc((size_t)L_ * D_ * D_ * 2);
  h16*   WvT    = (h16*)  alloc((size_t)L_ * D_ * D_ * 2);
  h16*   WoT    = (h16*)  alloc((size_t)L_ * D_ * D_ * 2);
  h16*   W1T    = (h16*)  alloc((size_t)L_ * D_ * DFF_ * 2);
  h16*   W2T    = (h16*)  alloc((size_t)L_ * DFF_ * D_ * 2);
  h16*   WoutT  = (h16*)  alloc((size_t)F_ * D_ * 2);
  (void)ws_size; (void)in_sizes; (void)n_in; (void)out_size;

  float* pred = (float*)d_out;                    // [B,S,F]
  float* fin  = (float*)d_out + (size_t)NR_ * F_; // [B,D/4]

  const int TB = 256;
  auto blocks = [](size_t n) { return (unsigned)((n + 255) / 256); };

  // ---- weight conversion (f32 -> transposed f16) ----
  tconv<<<blocks((size_t)F_ * D_), TB, 0, stream>>>(Wp, WpT, F_, D_);
  for (int l = 0; l < L_; ++l) {
    size_t w0 = (size_t)l * D_ * D_;
    tconv<<<blocks((size_t)D_ * D_), TB, 0, stream>>>(Wq + w0, WqT + w0, D_, D_);
    tconv<<<blocks((size_t)D_ * D_), TB, 0, stream>>>(Wk + w0, WkT + w0, D_, D_);
    tconv<<<blocks((size_t)D_ * D_), TB, 0, stream>>>(Wv + w0, WvT + w0, D_, D_);
    tconv<<<blocks((size_t)D_ * D_), TB, 0, stream>>>(Wo + w0, WoT + w0, D_, D_);
    size_t f0 = (size_t)l * D_ * DFF_;
    tconv<<<blocks((size_t)D_ * DFF_), TB, 0, stream>>>(W1 + f0, W1T + f0, D_, DFF_);
    tconv<<<blocks((size_t)DFF_ * D_), TB, 0, stream>>>(W2 + f0, W2T + f0, DFF_, D_);
  }
  tconv<<<blocks((size_t)D_ * F_), TB, 0, stream>>>(Wout, WoutT, D_, F_);

  // ---- input projection: src = (X @ Wp + bp) * sqrt(D) ----
  prep_x<<<blocks((size_t)B_ * S_ * F_), TB, 0, stream>>>(X, x16);
  launch_gemm(x16, WpT, bp, src32, src16, NR_, D_, F_, 0, 22.62741699796952f, stream);

  // ---- transformer layers ----
  for (int l = 0; l < L_; ++l) {
    size_t w0 = (size_t)l * D_ * D_;
    size_t f0 = (size_t)l * D_ * DFF_;
    launch_gemm(src16, WqT + w0, bq + l * D_, nullptr, q16,  NR_, D_, D_, 1, 1.f, stream);
    launch_gemm(src16, WkT + w0, bk + l * D_, nullptr, k16,  NR_, D_, D_, 1, 1.f, stream);
    launch_gemm(src16, WvT + w0, bv + l * D_, nullptr, v16v, NR_, D_, D_, 1, 1.f, stream);

    rope_heads<<<blocks((size_t)B_ * H_ * S_ * (HD_/2)), TB, 0, stream>>>(q16, Qh);
    rope_heads<<<blocks((size_t)B_ * H_ * S_ * (HD_/2)), TB, 0, stream>>>(k16, Kh);
    v_split_t <<<blocks((size_t)B_ * H_ * S_ * HD_),     TB, 0, stream>>>(v16v, Vt);

    attn_fused<<<dim3(S_ / 32, B_ * H_), TB, 0, stream>>>(Qh, Kh, Vt, pm, y16);

    launch_gemm(y16, WoT + w0, bo + l * D_, t32, nullptr, NR_, D_, D_, 1, 1.f, stream);
    add_ln<<<NR_, TB, 0, stream>>>(src32, t32, ln1g + l * D_, ln1b + l * D_, src16);

    launch_gemm(src16, W1T + f0, b1 + l * DFF_, nullptr, ff16, NR_, DFF_, D_, 2, 1.f, stream);
    launch_gemm(ff16,  W2T + f0, b2 + l * D_,   t32, nullptr,  NR_, D_, DFF_, 0, 1.f, stream);
    add_ln<<<NR_, TB, 0, stream>>>(src32, t32, ln2g + l * D_, ln2b + l * D_, src16);
  }

  // ---- heads ----
  gelu_permute<<<blocks((size_t)NR_ * D_), TB, 0, stream>>>(src32, emb32, emb16);
  launch_gemm(emb16, WoutT, bout, pred, nullptr, NR_, F_, D_, 0, 1.f, stream);

  pool_score  <<<blocks((size_t)NR_), TB, 0, stream>>>(emb32, Wpool, bpool, pscore);
  pool_combine<<<B_, TB, 0, stream>>>(pscore, emb32, pooled);
  proj_k <<<blocks((size_t)B_ * (D_/2)), TB, 0, stream>>>(pooled, Wproj, bproj, projb);
  bn_relu<<<1, D_/2, 0, stream>>>(projb, bng, bnb, projbn);
  fin_k  <<<blocks((size_t)B_ * (D_/4)), TB, 0, stream>>>(projbn, Wfin, bfin, fin);
}
